// TransformerSeqLayer_1881195676196
// MI455X (gfx1250) — matrix operand order repaired and checked by reading, not compiled, not run on hardware
//
#include <hip/hip_runtime.h>
#include <hip/hip_bf16.h>
#include <math.h>

#define B_    8
#define M_    1024
#define H_    512
#define K_    8
#define D_    64
#define SPAN_ 1024
#define MK_   2048

typedef __attribute__((ext_vector_type(16))) _Float16     v16h;
typedef __attribute__((ext_vector_type(8)))  float        v8f;
typedef __attribute__((ext_vector_type(4)))  unsigned int u32x4;

union Frag {
  v16h  v;
  u32x4 u[2];
};

__device__ __forceinline__ v8f wmma16(v16h a, v16h b, v8f c) {
  // D = A(16x32 f16) * B(32x16 f16) + C(16x16 f32)
  return __builtin_amdgcn_wmma_f32_16x16x32_f16(false, a, false, b, (short)0, c,
                                                false, false);
}

__device__ __forceinline__ v8f zero8() {
  v8f z = {0.f, 0.f, 0.f, 0.f, 0.f, 0.f, 0.f, 0.f};
  return z;
}

// A fragment 16x32: lane (r = lane&15, hi = lane>>4) holds row r,
// K = kk0 + hi*16 .. +15  (32B contiguous)
__device__ __forceinline__ v16h load_a_frag(const _Float16* base, int ld,
                                            int r, int hi, int kk0) {
  Frag f;
  const _Float16* p = base + (size_t)r * ld + kk0 + hi * 16;
  f.u[0] = *(const u32x4*)(p);
  f.u[1] = *(const u32x4*)(p + 8);
  return f.v;
}

// B fragment 32x16 from a buffer where column n of B == base[n*ld + kk]:
// lane (r, hi) holds column r, K = kk0 + hi*16 .. +15  (32B contiguous)
__device__ __forceinline__ v16h load_b_frag(const _Float16* base, int ld,
                                            int r, int hi, int kk0) {
  Frag f;
  const _Float16* p = base + (size_t)r * ld + kk0 + hi * 16;
  f.u[0] = *(const u32x4*)(p);
  f.u[1] = *(const u32x4*)(p + 8);
  return f.v;
}

// ---------------------------------------------------------------- converts
__global__ void cast_f32_to_f16(const float* __restrict__ in,
                                _Float16* __restrict__ out, int n) {
  int i = blockIdx.x * blockDim.x + threadIdx.x;
  if (i < n) out[i] = (_Float16)in[i];
}

// out[c*rows + r] = in[r*cols + c]  (store transposed, f16)
__global__ void transpose_f32_to_f16(const float* __restrict__ in,
                                     _Float16* __restrict__ out,
                                     int rows, int cols) {
  int i = blockIdx.x * blockDim.x + threadIdx.x;
  if (i < rows * cols) {
    int rr = i / cols;
    int cc = i - rr * cols;
    out[(size_t)cc * rows + rr] = (_Float16)in[i];
  }
}

// ------------------------------------------------- projection GEMM (x @ W)
// X: [T][512] f16 row-major. WT: [512 out][512 in] f16 (W transposed).
// mode 0: OUT = Q  [B*K][M][64]     (rows_shift = 10)
// mode 1: OUT = K  [B*K][MK][64]    (rows_shift = 11)
// mode 2: OUT = vT [B*K][64][MK]    (rows_shift = 11)
__global__ void proj_gemm_f16(const _Float16* __restrict__ X,
                              const _Float16* __restrict__ WT,
                              _Float16* __restrict__ OUT,
                              int rows_shift, int mode) {
  const int lane = threadIdx.x & 31;
  const int wave = threadIdx.x >> 5;
  const int r    = lane & 15;
  const int hi   = lane >> 4;
  const int row0 = blockIdx.x * 16;
  const int n0   = blockIdx.y * 128 + wave * 16;

  const _Float16* Abase = X + (size_t)row0 * H_;
  const _Float16* Bbase = WT + (size_t)n0 * H_;

  v8f acc = zero8();
#pragma unroll
  for (int kc = 0; kc < H_ / 32; ++kc) {
    v16h a = load_a_frag(Abase, H_, r, hi, kc * 32);
    v16h b = load_b_frag(Bbase, H_, r, hi, kc * 32);
    acc = wmma16(a, b, acc);
  }

  const int rows_pb = 1 << rows_shift;
  const int col = n0 + r;
  const int h   = col >> 6;
  const int d   = col & 63;
#pragma unroll
  for (int vr = 0; vr < 8; ++vr) {
    int grow = row0 + vr + 8 * hi;
    int bb   = grow >> rows_shift;
    int t    = grow & (rows_pb - 1);
    size_t idx;
    if (mode == 2)
      idx = ((size_t)(bb * K_ + h) * D_ + d) * MK_ + t;          // vT layout
    else
      idx = ((size_t)(bb * K_ + h) * rows_pb + t) * (size_t)D_ + d;
    OUT[idx] = (_Float16)acc[vr];
  }
}

// ------------------------------------------------- final GEMM (ctx @ Wo)
__global__ void out_gemm_f32(const _Float16* __restrict__ X,
                             const _Float16* __restrict__ WT,
                             float* __restrict__ OUT) {
  const int lane = threadIdx.x & 31;
  const int wave = threadIdx.x >> 5;
  const int r    = lane & 15;
  const int hi   = lane >> 4;
  const int row0 = blockIdx.x * 16;
  const int n0   = blockIdx.y * 128 + wave * 16;

  const _Float16* Abase = X + (size_t)row0 * H_;
  const _Float16* Bbase = WT + (size_t)n0 * H_;

  v8f acc = zero8();
#pragma unroll
  for (int kc = 0; kc < H_ / 32; ++kc) {
    v16h a = load_a_frag(Abase, H_, r, hi, kc * 32);
    v16h b = load_b_frag(Bbase, H_, r, hi, kc * 32);
    acc = wmma16(a, b, acc);
  }
#pragma unroll
  for (int vr = 0; vr < 8; ++vr) {
    int grow = row0 + vr + 8 * hi;
    OUT[(size_t)grow * H_ + n0 + r] = acc[vr];
  }
}

// ------------------------------------------------- banded attention kernel
// One block = one head (blockIdx.y) x one 16-query tile (blockIdx.x), 256 thr.
// LDS: s_sc  f32[16][1056]  skewed raw scores (col = l - m0)
//      s_at  f16[16][1056]  skewed softmax weights, zero-padded outside band
//      s_red f32[16][16]    softmax reduction scratch
#define SKW_ 1056
__global__ void attn_kernel(const _Float16* __restrict__ Q,
                            const _Float16* __restrict__ KH,
                            const _Float16* __restrict__ VT,
                            const _Float16* __restrict__ POST,
                            _Float16* __restrict__ CTX) {
  extern __shared__ char smem[];
  float*    s_sc  = (float*)smem;
  _Float16* s_at  = (_Float16*)(smem + 16 * SKW_ * 4);
  float*    s_red = (float*)(smem + 16 * SKW_ * 4 + 16 * SKW_ * 2);

  const int tid  = threadIdx.x;
  const int lane = tid & 31;
  const int wave = tid >> 5;
  const int r    = lane & 15;
  const int hi   = lane >> 4;

  const int m0   = blockIdx.x * 16;
  const int head = blockIdx.y;
  const int bb   = head >> 3;   // batch
  const int hh   = head & 7;    // head-in-batch

  const _Float16* qh = Q  + (size_t)head * M_ * D_;
  const _Float16* kh = KH + (size_t)head * MK_ * D_;
  const _Float16* vt = VT + (size_t)head * D_ * MK_;

  // zero skewed attention buffer (pads must be 0 for phase 3)
  for (int i = tid; i < 16 * SKW_; i += 256) s_at[i] = (_Float16)0.f;

  // Q A-fragments for this tile (K = D = 64 -> two frags), reused everywhere
  const _Float16* qb = qh + (size_t)m0 * D_;
  v16h a_lo = load_a_frag(qb, D_, r, hi, 0);
  v16h a_hi = load_a_frag(qb, D_, r, hi, 32);

  // phase 1a: ctx scores q.kT in skewed coords; 66 l-tiles cover l-m0 in [0,1056)
  for (int t = wave; t < 66; t += 8) {
    int l0   = m0 + t * 16;
    int krow = l0 + r;
    if (krow > MK_ - 1) krow = MK_ - 1;   // out-of-band tail, never read
    const _Float16* kp = kh + (size_t)krow * D_;
    Frag b0, b1;
    b0.u[0] = *(const u32x4*)(kp + hi * 16);
    b0.u[1] = *(const u32x4*)(kp + hi * 16 + 8);
    b1.u[0] = *(const u32x4*)(kp + 32 + hi * 16);
    b1.u[1] = *(const u32x4*)(kp + 32 + hi * 16 + 8);
    v8f acc = zero8();
    acc = wmma16(a_lo, b0.v, acc);
    acc = wmma16(a_hi, b1.v, acc);
#pragma unroll
    for (int vr = 0; vr < 8; ++vr) {
      int row = vr + 8 * hi;
      s_sc[row * SKW_ + t * 16 + r] = acc[vr] * 0.125f;   // 1/sqrt(D)
    }
  }
  __syncthreads();

  // phase 1b: positional scores q.posT (unskewed s), added at skewed offsets
  for (int st = wave * 8; st < wave * 8 + 8; ++st) {
    int s0 = st * 16;
    v16h pb0 = load_b_frag(POST + (size_t)s0 * D_, D_, r, hi, 0);
    v16h pb1 = load_b_frag(POST + (size_t)s0 * D_, D_, r, hi, 32);
    v8f acc = zero8();
    acc = wmma16(a_lo, pb0, acc);
    acc = wmma16(a_hi, pb1, acc);
#pragma unroll
    for (int vr = 0; vr < 8; ++vr) {
      int row = vr + 8 * hi;
      atomicAdd(&s_sc[row * SKW_ + s0 + r + row], acc[vr] * 0.125f);
    }
  }
  __syncthreads();

  // phase 2: softmax over s in [0,1024); 16 threads per row
  {
    int sr  = tid >> 4;
    int t16 = tid & 15;
    float mx = -3.0e38f;
    for (int j = 0; j < 64; ++j)
      mx = fmaxf(mx, s_sc[sr * SKW_ + (t16 + 16 * j) + sr]);
    s_red[sr * 16 + t16] = mx;
    __syncthreads();
    float rmax = s_red[sr * 16];
    for (int i = 1; i < 16; ++i) rmax = fmaxf(rmax, s_red[sr * 16 + i]);
    __syncthreads();
    float sm = 0.f;
    for (int j = 0; j < 64; ++j) {
      int c = (t16 + 16 * j) + sr;
      float e = __expf(s_sc[sr * SKW_ + c] - rmax);
      s_sc[sr * SKW_ + c] = e;
      sm += e;
    }
    s_red[sr * 16 + t16] = sm;
    __syncthreads();
    float rsum = 0.f;
    for (int i = 0; i < 16; ++i) rsum += s_red[sr * 16 + i];
    float inv = 1.f / rsum;
    for (int j = 0; j < 64; ++j) {
      int c = (t16 + 16 * j) + sr;
      s_at[sr * SKW_ + c] = (_Float16)(s_sc[sr * SKW_ + c] * inv);
    }
  }
  __syncthreads();

  // phase 3: out = attn_sk @ v over 33 K=32 chunks; waves 0..3 own d-tiles
  if (wave < 4) {
    int d0 = wave * 16;
    const _Float16* vrow = vt + (size_t)(d0 + r) * MK_;
    v8f acc = zero8();
    for (int c = 0; c < 33; ++c) {
      v16h a = load_a_frag(s_at, SKW_, r, hi, c * 32);   // aligned ds_load_b128
      int s = m0 + c * 32 + hi * 16;
      if (s > MK_ - 16) s = MK_ - 16;   // tail: attn weights there are 0
      Frag b;
      b.u[0] = *(const u32x4*)(vrow + s);
      b.u[1] = *(const u32x4*)(vrow + s + 8);
      acc = wmma16(a, b.v, acc);
    }
#pragma unroll
    for (int vr = 0; vr < 8; ++vr) {
      int grow = bb * M_ + m0 + vr + 8 * hi;
      int col  = hh * 64 + d0 + r;
      CTX[(size_t)grow * H_ + col] = (_Float16)acc[vr];
    }
  }
}

// -------------------------------------------------------------- launcher
extern "C" void kernel_launch(void* const* d_in, const int* in_sizes, int n_in,
                              void* d_out, int out_size, void* d_ws, size_t ws_size,
                              hipStream_t stream) {
  const float* q_in   = (const float*)d_in[0];
  const float* k_in   = (const float*)d_in[1];
  const float* v_in   = (const float*)d_in[2];
  const float* pos_in = (const float*)d_in[3];
  const float* Wq     = (const float*)d_in[4];
  const float* Wk     = (const float*)d_in[5];
  const float* Wv     = (const float*)d_in[6];
  const float* Wo     = (const float*)d_in[7];
  float* out = (float*)d_out;

  _Float16* ws = (_Float16*)d_ws;
  size_t off = 0;
  _Float16* WqT = ws + off; off += (size_t)H_ * H_;
  _Float16* WkT = ws + off; off += (size_t)H_ * H_;
  _Float16* WvT = ws + off; off += (size_t)H_ * H_;
  _Float16* WoT = ws + off; off += (size_t)H_ * H_;
  _Float16* PosT = ws + off; off += (size_t)SPAN_ * D_;
  _Float16* Xq = ws + off; off += (size_t)B_ * M_ * H_;
  _Float16* Xk = ws + off; off += (size_t)B_ * MK_ * H_;
  _Float16* Xv = ws + off; off += (size_t)B_ * MK_ * H_;
  _Float16* Qh = ws + off; off += (size_t)B_ * K_ * M_ * D_;
  _Float16* Kh = ws + off; off += (size_t)B_ * K_ * MK_ * D_;
  _Float16* Vt = ws + off; off += (size_t)B_ * K_ * MK_ * D_;
  _Float16* Ctx = ws + off; off += (size_t)B_ * M_ * H_;

  // converts (f32 -> f16; weights & pos transposed for contiguous B-frags)
  int nW = H_ * H_;
  transpose_f32_to_f16<<<(nW + 255) / 256, 256, 0, stream>>>(Wq, WqT, H_, H_);
  transpose_f32_to_f16<<<(nW + 255) / 256, 256, 0, stream>>>(Wk, WkT, H_, H_);
  transpose_f32_to_f16<<<(nW + 255) / 256, 256, 0, stream>>>(Wv, WvT, H_, H_);
  transpose_f32_to_f16<<<(nW + 255) / 256, 256, 0, stream>>>(Wo, WoT, H_, H_);
  int nP = D_ * SPAN_;
  transpose_f32_to_f16<<<(nP + 255) / 256, 256, 0, stream>>>(pos_in, PosT, D_, SPAN_);
  int nQ = B_ * M_ * H_, nK = B_ * MK_ * H_;
  cast_f32_to_f16<<<(nQ + 255) / 256, 256, 0, stream>>>(q_in, Xq, nQ);
  cast_f32_to_f16<<<(nK + 255) / 256, 256, 0, stream>>>(k_in, Xk, nK);
  cast_f32_to_f16<<<(nK + 255) / 256, 256, 0, stream>>>(v_in, Xv, nK);

  // projections (head-split; v transposed per head)
  proj_gemm_f16<<<dim3(B_ * M_ / 16, H_ / 128), 256, 0, stream>>>(Xq, WqT, Qh, 10, 0);
  proj_gemm_f16<<<dim3(B_ * MK_ / 16, H_ / 128), 256, 0, stream>>>(Xk, WkT, Kh, 11, 1);
  proj_gemm_f16<<<dim3(B_ * MK_ / 16, H_ / 128), 256, 0, stream>>>(Xv, WvT, Vt, 11, 2);

  // banded attention + softmax + attn@v
  int smem = 16 * SKW_ * 4 + 16 * SKW_ * 2 + 16 * 16 * 4;   // 102400 B
  hipFuncSetAttribute(reinterpret_cast<const void*>(attn_kernel),
                      hipFuncAttributeMaxDynamicSharedMemorySize, smem);
  attn_kernel<<<dim3(M_ / 16, B_ * K_), 256, smem, stream>>>(Qh, Kh, Vt, PosT, Ctx);

  // output projection
  out_gemm_f32<<<dim3(B_ * M_ / 16, H_ / 128), 256, 0, stream>>>(Ctx, WoT, out);
}
